// Attention_87024627351803
// MI455X (gfx1250) — matrix operand order repaired and checked by reading, not compiled, not run on hardware
//
#include <hip/hip_runtime.h>
#include <hip/hip_bf16.h>

// ---------------------------------------------------------------------------
// Problem constants
// ---------------------------------------------------------------------------
#define BATCH     2
#define CCH       256
#define TT        4096          // H*W = 64*64
#define NHEADS    4
#define HEADS_TOT (BATCH * NHEADS)   // 8
#define CH_HEAD   64
#define GROUPS    32
#define GROUP_SZ  (8 * TT)      // (C/GROUPS) * T = 32768
// softmax scale: 1/sqrt(sqrt(64)) applied to both q and k (as in reference)
#define ATT_SCALE 0.35355339059327373f

typedef __attribute__((ext_vector_type(16))) __bf16 v16bf;
typedef __attribute__((ext_vector_type(8)))  float  v8f;

#define LDA_S  48   // LDS ld (elems) for 64x32 A tile: 96 B/row, 16B aligned
#define LDBT_S 40   // LDS ld (elems) for 64x32 Bt tile: 80 B/row, 16B aligned
#define LDQ    72   // LDS ld (elems) for attention 64x64 tiles: 144 B/row

__device__ __forceinline__ v8f zero8() {
  v8f z;
#pragma unroll
  for (int i = 0; i < 8; ++i) z[i] = 0.0f;
  return z;
}

// ---------------------------------------------------------------------------
// WMMA fragment builders (CDNA5 ISA 7.12.2, wave32), fully vectorized:
//  A (16x32 bf16), tile row-major [M][K]:
//    lane m = lane&15, half = lane>>4; elems j=0..7  -> k = half*8 + j
//                                     elems j=8..15 -> k = 16 + half*8 + (j-8)
//    => two contiguous b128 loads at r+half*8 and r+16+half*8.
//  B (32x16 bf16), tile stored TRANSPOSED [N][K] ("Bt"):
//    lane n = lane&15; same k mapping as A => two b128 loads at half*8, 16+half*8.
//  C/D (16x16 f32): col n = lane&15; elem i -> row m = half*8 + i
// ---------------------------------------------------------------------------
__device__ __forceinline__ v16bf ld_frag_a(const __bf16* tile, int row, int ld,
                                           int koff) {
  const int lane = threadIdx.x & 31;
  const int half = (lane >> 4) & 1;
  const __bf16* r = tile + (size_t)row * ld + koff + half * 8;
  union { uint4 u[2]; v16bf v; } c;
  c.u[0] = *(const uint4*)r;
  c.u[1] = *(const uint4*)(r + 16);
  return c.v;
}

__device__ __forceinline__ v16bf ld_frag_bt(const __bf16* tile, int col, int ld,
                                            int koff) {
  const int lane = threadIdx.x & 31;
  const int half = (lane >> 4) & 1;
  const __bf16* r =
      tile + (size_t)(col + (lane & 15)) * ld + koff + half * 8;
  union { uint4 u[2]; v16bf v; } c;
  c.u[0] = *(const uint4*)r;
  c.u[1] = *(const uint4*)(r + 16);
  return c.v;
}

#define WMMA_BF16(A, B, C)                                                  \
  __builtin_amdgcn_wmma_f32_16x16x32_bf16(false, (A), false, (B), (short)0, \
                                          (C), false, false)

// ---------------------------------------------------------------------------
// Shared 64x64 GEMM tile engine: block = 128 threads = 4 waves.
// Wave w computes rows [m0+16w, m0+16w+16) x cols [n0, n0+64).
// A : row-major MxK bf16 (lda).   BT: row-major NxK bf16 (ldbt) — i.e. the
// mathematical B stored transposed, so LDS staging and fragment loads are all
// 128-bit.
// ---------------------------------------------------------------------------
__device__ __forceinline__ void gemm_tile(const __bf16* __restrict__ A, int lda,
                                          int m0, const __bf16* __restrict__ BT,
                                          int ldbt, int n0, int KTOT,
                                          __bf16* Asub, __bf16* Bts,
                                          v8f acc[4]) {
  const int tid = threadIdx.x;
  const int w = tid >> 5;
  const int lane = tid & 31;
  const int nidx = lane & 15;
#pragma unroll
  for (int nt = 0; nt < 4; ++nt) acc[nt] = zero8();

  for (int kt = 0; kt < KTOT; kt += 32) {
    {  // stage A tile 64x32 (2 threads per row, 16 bf16 = 2x b128 each)
      const int row = tid >> 1, seg = tid & 1;
      const uint4* s4 =
          (const uint4*)(A + (size_t)(m0 + row) * lda + kt + seg * 16);
      uint4* d = (uint4*)(Asub + row * LDA_S + seg * 16);
      d[0] = s4[0];
      d[1] = s4[1];
    }
    {  // stage Bt tile 64(n)x32(k), all 128-bit
      const int row = tid >> 1, seg = tid & 1;
      const uint4* s4 =
          (const uint4*)(BT + (size_t)(n0 + row) * ldbt + kt + seg * 16);
      uint4* d = (uint4*)(Bts + row * LDBT_S + seg * 16);
      d[0] = s4[0];
      d[1] = s4[1];
    }
    if (kt + 32 < KTOT)  // -> global_prefetch_b8
      __builtin_prefetch(
          BT + (size_t)(n0 + (tid >> 1)) * ldbt + kt + 32 + (tid & 1) * 16, 0,
          1);
    __syncthreads();
    const v16bf af = ld_frag_a(Asub, w * 16 + nidx, LDA_S, 0);
#pragma unroll
    for (int nt = 0; nt < 4; ++nt) {
      const v16bf bf = ld_frag_bt(Bts, nt * 16, LDBT_S, 0);
      acc[nt] = WMMA_BF16(af, bf, acc[nt]);
    }
    __syncthreads();
  }
}

// ---------------------------------------------------------------------------
// Kernel 1: GroupNorm statistics (one block per (batch, group))
// ---------------------------------------------------------------------------
__global__ void gn_stats(const float* __restrict__ x, float* __restrict__ mean,
                         float* __restrict__ rstd) {
  __shared__ float sh[512];
  const int g = blockIdx.x;  // 0..63 = b*32 + group (channels contiguous)
  const float* p = x + (size_t)g * GROUP_SZ;
  float s = 0.f, ss = 0.f;
  for (int i = threadIdx.x; i < GROUP_SZ; i += 256) {
    const float v = p[i];
    s += v;
    ss += v * v;
  }
  sh[threadIdx.x] = s;
  sh[256 + threadIdx.x] = ss;
  __syncthreads();
  for (int st = 128; st > 0; st >>= 1) {
    if ((int)threadIdx.x < st) {
      sh[threadIdx.x] += sh[threadIdx.x + st];
      sh[256 + threadIdx.x] += sh[256 + threadIdx.x + st];
    }
    __syncthreads();
  }
  if (threadIdx.x == 0) {
    const float mu = sh[0] * (1.0f / GROUP_SZ);
    const float var = sh[256] * (1.0f / GROUP_SZ) - mu * mu;
    mean[g] = mu;
    rstd[g] = rsqrtf(var + 1e-5f);
  }
}

// ---------------------------------------------------------------------------
// Kernel 2: apply GroupNorm affine, cast to bf16 AND transpose to [t][c]
// via a 64x64 LDS tile (all global traffic coalesced/vectorized).
// ---------------------------------------------------------------------------
__global__ void __launch_bounds__(256) gn_apply_t(
    const float* __restrict__ x, const float* __restrict__ mean,
    const float* __restrict__ rstd, const float* __restrict__ nw,
    const float* __restrict__ nb, __bf16* __restrict__ xnT) {
  __shared__ __bf16 Ts[64 * LDQ];  // [t_local][c_local]
  const int t0 = blockIdx.x * 64;
  const int c0 = blockIdx.y * 64;
  const int b = blockIdx.z;
  const int tid = threadIdx.x;
  const int ci = tid >> 2, seg = tid & 3;
  const int c = c0 + ci;
  const int g = b * GROUPS + (c >> 3);
  const float mu = mean[g], rs = rstd[g];
  const float wgt = nw[c], bia = nb[c];
  const float* src = x + ((size_t)b * CCH + c) * TT + t0 + seg * 16;
#pragma unroll
  for (int i = 0; i < 16; ++i) {
    const float v = (src[i] - mu) * rs * wgt + bia;
    Ts[(seg * 16 + i) * LDQ + ci] = (__bf16)v;  // LDS-side transpose
  }
  __syncthreads();
  const int row = tid >> 2;  // t row 0..63
  const uint4* s4 = (const uint4*)(Ts + row * LDQ + seg * 16);
  uint4* d = (uint4*)(xnT + ((size_t)b * TT + t0 + row) * CCH + c0 + seg * 16);
  d[0] = s4[0];
  d[1] = s4[1];
}

// ---------------------------------------------------------------------------
// Kernel 3: cast GEMM weights to bf16
// ---------------------------------------------------------------------------
__global__ void cast_w(const float* __restrict__ qkv_w,
                       const float* __restrict__ proj_w,
                       __bf16* __restrict__ wq, __bf16* __restrict__ wp) {
  const int idx = blockIdx.x * 256 + threadIdx.x;
  if (idx < 768 * 256) wq[idx] = (__bf16)qkv_w[idx];
  if (idx < 256 * 256) wp[idx] = (__bf16)proj_w[idx];
}

// ---------------------------------------------------------------------------
// Kernel 4: QKV GEMM (768x4096x256 per batch) + bias + head scatter + scale.
// Outputs: Qt,Kt as [head][t][c] (row-major A / Bt-ready for attention),
//          V  as  [head][c][t]  (Bt-ready for the P*V^T GEMM).
// ---------------------------------------------------------------------------
__global__ void __launch_bounds__(128) qkv_gemm(
    const __bf16* __restrict__ wq, const __bf16* __restrict__ xnT,
    const float* __restrict__ qkv_b, __bf16* __restrict__ Qt,
    __bf16* __restrict__ Kt, __bf16* __restrict__ Vb) {
  __shared__ __bf16 Asub[64 * LDA_S];
  __shared__ __bf16 Bts[64 * LDBT_S];
  const int m0 = blockIdx.x * 64;
  const int n0 = blockIdx.y * 64;
  const int b = blockIdx.z;
  const __bf16* BT = xnT + (size_t)b * TT * CCH;
  v8f acc[4];
  gemm_tile(wq, CCH, m0, BT, CCH, n0, CCH, Asub, Bts, acc);

  const int lane = threadIdx.x & 31;
  const int half = (lane >> 4) & 1;
  const int nidx = lane & 15;
  const int w = threadIdx.x >> 5;
#pragma unroll
  for (int nt = 0; nt < 4; ++nt) {
    const int t = n0 + nt * 16 + nidx;
#pragma unroll
    for (int i = 0; i < 8; ++i) {
      const int o = m0 + w * 16 + half * 8 + i;  // output row 0..767
      const float v = acc[nt][i] + qkv_b[o];
      const int hh = o / 192;  // head within batch (legacy split)
      const int j = o % 192;   // 0..63 q, 64..127 k, 128..191 v
      const int jj = j & 63;
      const int bh = b * NHEADS + hh;
      if (j < 64)
        Qt[((size_t)bh * TT + t) * CH_HEAD + jj] = (__bf16)(v * ATT_SCALE);
      else if (j < 128)
        Kt[((size_t)bh * TT + t) * CH_HEAD + jj] = (__bf16)(v * ATT_SCALE);
      else
        Vb[((size_t)bh * CH_HEAD + jj) * TT + t] = (__bf16)v;
    }
  }
}

// ---------------------------------------------------------------------------
// Kernel 5: flash attention. Block = 128 thr = 4 waves; 1 block per
// (64-query tile, head). Wave w owns 16 query rows. Streams K/V tiles of 64
// keys through LDS; online softmax; O accumulated in registers.
// All staging copies and fragment loads are 128-bit.
// ---------------------------------------------------------------------------
__global__ void __launch_bounds__(128) flash_attn(
    const __bf16* __restrict__ Qt, const __bf16* __restrict__ Kt,
    const __bf16* __restrict__ Vb, __bf16* __restrict__ AbT) {
  __shared__ __bf16 Qs[64 * LDQ];     // [t_local][c]   (A layout)
  __shared__ __bf16 Ks[64 * LDQ];     // [s_local][c]   (Bt layout, k = c)
  __shared__ __bf16 Vs[64 * LDQ];     // [c][s_local]   (Bt layout, k = s)
  __shared__ __bf16 Ps[4][16 * LDQ];  // per-wave P tile [q_local16][s]

  const int qb = blockIdx.x;  // 0..63
  const int h = blockIdx.y;   // 0..7 (= b*4 + head)
  const int tid = threadIdx.x;
  const int w = tid >> 5;
  const int lane = tid & 31;
  const int half = (lane >> 4) & 1;
  const int nidx = lane & 15;
  const int t0 = qb * 64;

  {  // Q tile: rows of Qt are already [t][c]
    const int tl = tid >> 1, seg = tid & 1;
    const uint4* s4 =
        (const uint4*)(Qt + ((size_t)h * TT + t0 + tl) * CH_HEAD + seg * 32);
    uint4* d = (uint4*)(Qs + tl * LDQ + seg * 32);
    d[0] = s4[0]; d[1] = s4[1]; d[2] = s4[2]; d[3] = s4[3];
  }

  float m_i[8], l_i[8];
  v8f o_acc[4];
#pragma unroll
  for (int i = 0; i < 8; ++i) { m_i[i] = -1e30f; l_i[i] = 0.0f; }
#pragma unroll
  for (int ct = 0; ct < 4; ++ct) o_acc[ct] = zero8();

  for (int s0 = 0; s0 < TT; s0 += 64) {
    __syncthreads();  // protect Qs (first iter) and Ks/Vs reuse
    {  // K tile: [s_local][c]
      const int sl = tid >> 1, seg = tid & 1;
      const uint4* s4 =
          (const uint4*)(Kt + ((size_t)h * TT + s0 + sl) * CH_HEAD + seg * 32);
      uint4* d = (uint4*)(Ks + sl * LDQ + seg * 32);
      d[0] = s4[0]; d[1] = s4[1]; d[2] = s4[2]; d[3] = s4[3];
    }
    {  // V tile: [c][s_local]
      const int c = tid >> 1, seg = tid & 1;
      const uint4* s4 =
          (const uint4*)(Vb + ((size_t)h * CH_HEAD + c) * TT + s0 + seg * 32);
      uint4* d = (uint4*)(Vs + c * LDQ + seg * 32);
      d[0] = s4[0]; d[1] = s4[1]; d[2] = s4[2]; d[3] = s4[3];
    }
    __syncthreads();

    // --- S = (Q*scale)^T (K*scale): 16x64 per wave, K-dim = ch 64 ---
    v8f s_acc[4];
#pragma unroll
    for (int nt = 0; nt < 4; ++nt) s_acc[nt] = zero8();
#pragma unroll
    for (int koff = 0; koff < 64; koff += 32) {
      const v16bf aq = ld_frag_a(Qs, w * 16 + nidx, LDQ, koff);
#pragma unroll
      for (int nt = 0; nt < 4; ++nt) {
        const v16bf bk = ld_frag_bt(Ks, nt * 16, LDQ, koff);
        s_acc[nt] = WMMA_BF16(aq, bk, s_acc[nt]);
      }
    }

    // --- online softmax: rows live in (half, vgpr-slot); cols across lanes
    float rs[8];
#pragma unroll
    for (int i = 0; i < 8; ++i) {
      float mx = fmaxf(fmaxf(s_acc[0][i], s_acc[1][i]),
                       fmaxf(s_acc[2][i], s_acc[3][i]));
      mx = fmaxf(mx, __shfl_xor(mx, 1, 32));
      mx = fmaxf(mx, __shfl_xor(mx, 2, 32));
      mx = fmaxf(mx, __shfl_xor(mx, 4, 32));
      mx = fmaxf(mx, __shfl_xor(mx, 8, 32));
      const float mi = fmaxf(m_i[i], mx);
      const float alpha = __expf(m_i[i] - mi);
      m_i[i] = mi;
      l_i[i] *= alpha;
#pragma unroll
      for (int nt = 0; nt < 4; ++nt) s_acc[nt][i] = __expf(s_acc[nt][i] - mi);
      rs[i] = s_acc[0][i] + s_acc[1][i] + s_acc[2][i] + s_acc[3][i];
#pragma unroll
      for (int ct = 0; ct < 4; ++ct) o_acc[ct][i] *= alpha;
    }
#pragma unroll
    for (int i = 0; i < 8; ++i) {
      float r = rs[i];
      r += __shfl_xor(r, 1, 32);
      r += __shfl_xor(r, 2, 32);
      r += __shfl_xor(r, 4, 32);
      r += __shfl_xor(r, 8, 32);
      l_i[i] += r;
    }

    // --- stage P (bf16) into per-wave LDS tile [q16][s64] ---
#pragma unroll
    for (int nt = 0; nt < 4; ++nt)
#pragma unroll
      for (int i = 0; i < 8; ++i)
        Ps[w][(half * 8 + i) * LDQ + nt * 16 + nidx] = (__bf16)s_acc[nt][i];
    // same-wave LDS store -> load; wait all DS ops (CDNA5 split counters)
    asm volatile("s_wait_dscnt 0" ::: "memory");

    // --- O += P * V^T : K-dim = s 64, N = ch 64 ---
#pragma unroll
    for (int koff = 0; koff < 64; koff += 32) {
      const v16bf ap = ld_frag_a(&Ps[w][0], nidx, LDQ, koff);
#pragma unroll
      for (int ct = 0; ct < 4; ++ct) {
        const v16bf bv = ld_frag_bt(Vs, ct * 16, LDQ, koff);
        o_acc[ct] = WMMA_BF16(ap, bv, o_acc[ct]);
      }
    }
  }

  // epilogue: a^T[b][t][hh*64 + c] = O / l   (Bt-ready for proj GEMM)
  const int bb = h >> 2, hh = h & 3;
#pragma unroll
  for (int ct = 0; ct < 4; ++ct) {
    const int c = hh * CH_HEAD + ct * 16 + nidx;
#pragma unroll
    for (int i = 0; i < 8; ++i) {
      const int t = t0 + w * 16 + half * 8 + i;
      AbT[((size_t)bb * TT + t) * CCH + c] = (__bf16)(o_acc[ct][i] / l_i[i]);
    }
  }
}

// ---------------------------------------------------------------------------
// Kernel 6: projection GEMM (256x4096x256 per batch) + bias + residual (f32)
// ---------------------------------------------------------------------------
__global__ void __launch_bounds__(128) proj_gemm(
    const __bf16* __restrict__ wp, const __bf16* __restrict__ AbT,
    const float* __restrict__ proj_b, const float* __restrict__ x,
    float* __restrict__ out) {
  __shared__ __bf16 Asub[64 * LDA_S];
  __shared__ __bf16 Bts[64 * LDBT_S];
  const int m0 = blockIdx.x * 64;
  const int n0 = blockIdx.y * 64;
  const int b = blockIdx.z;
  const __bf16* BT = AbT + (size_t)b * TT * CCH;
  v8f acc[4];
  gemm_tile(wp, CCH, m0, BT, CCH, n0, CCH, Asub, Bts, acc);

  const int lane = threadIdx.x & 31;
  const int half = (lane >> 4) & 1;
  const int nidx = lane & 15;
  const int w = threadIdx.x >> 5;
#pragma unroll
  for (int nt = 0; nt < 4; ++nt) {
    const int t = n0 + nt * 16 + nidx;
#pragma unroll
    for (int i = 0; i < 8; ++i) {
      const int m = m0 + w * 16 + half * 8 + i;
      const size_t idx = ((size_t)b * CCH + m) * TT + t;
      out[idx] = x[idx] + acc[nt][i] + proj_b[m];
    }
  }
}

// ---------------------------------------------------------------------------
// Launch. ws layout (bytes):
//   [0,1024)  mean[64], rstd[64]
//   +1024     xnT  bf16 [b][t][c]        2*4096*256   (4 MB)
//   ...       wq bf16 768*256 ; wp bf16 256*256
//   ...       Qt,Kt bf16 [head][t][c] ; V bf16 [head][c][t]  (4 MB each)
//   ...       AbT bf16 [b][t][c]                       (4 MB)   ~20.6 MB total
// ---------------------------------------------------------------------------
extern "C" void kernel_launch(void* const* d_in, const int* in_sizes, int n_in,
                              void* d_out, int out_size, void* d_ws,
                              size_t ws_size, hipStream_t stream) {
  const float* x      = (const float*)d_in[0];
  const float* norm_w = (const float*)d_in[1];
  const float* norm_b = (const float*)d_in[2];
  const float* qkv_w  = (const float*)d_in[3];
  const float* qkv_b  = (const float*)d_in[4];
  const float* proj_w = (const float*)d_in[5];
  const float* proj_b = (const float*)d_in[6];
  float* out = (float*)d_out;

  char* ws = (char*)d_ws;
  float* meanp = (float*)ws;
  float* rstdp = meanp + 64;
  size_t off = 1024;
  __bf16* xnT = (__bf16*)(ws + off); off += (size_t)BATCH * CCH * TT * 2;
  __bf16* wq  = (__bf16*)(ws + off); off += (size_t)768 * 256 * 2;
  __bf16* wp  = (__bf16*)(ws + off); off += (size_t)256 * 256 * 2;
  __bf16* Qt  = (__bf16*)(ws + off); off += (size_t)HEADS_TOT * CH_HEAD * TT * 2;
  __bf16* Kt  = (__bf16*)(ws + off); off += (size_t)HEADS_TOT * CH_HEAD * TT * 2;
  __bf16* Vb  = (__bf16*)(ws + off); off += (size_t)HEADS_TOT * CH_HEAD * TT * 2;
  __bf16* AbT = (__bf16*)(ws + off);

  gn_stats<<<BATCH * GROUPS, 256, 0, stream>>>(x, meanp, rstdp);
  gn_apply_t<<<dim3(TT / 64, CCH / 64, BATCH), 256, 0, stream>>>(
      x, meanp, rstdp, norm_w, norm_b, xnT);
  cast_w<<<(768 * 256) / 256, 256, 0, stream>>>(qkv_w, proj_w, wq, wp);
  qkv_gemm<<<dim3(768 / 64, TT / 64, BATCH), 128, 0, stream>>>(wq, xnT, qkv_b,
                                                               Qt, Kt, Vb);
  flash_attn<<<dim3(TT / 64, HEADS_TOT), 128, 0, stream>>>(Qt, Kt, Vb, AbT);
  proj_gemm<<<dim3(CCH / 64, TT / 64, BATCH), 128, 0, stream>>>(wp, AbT, proj_b,
                                                                x, out);
}
